// GraphTransformerAutoencoderAlibi_12661563588873
// MI455X (gfx1250) — compile-verified, matrix operand order checked by reading, not run on hardware
//
#include <hip/hip_runtime.h>
#include <math.h>

typedef __attribute__((ext_vector_type(16))) _Float16 v16h;
typedef __attribute__((ext_vector_type(8)))  _Float16 v8h;
typedef __attribute__((ext_vector_type(4)))  _Float16 v4h;
typedef __attribute__((ext_vector_type(8)))  float    v8f;

#define TB   2048              // batch
#define TD   1024              // model dim
#define TNN  6                 // nodes (tokens)
#define THH  4                 // heads
#define TDH  256               // head dim
#define TM   (TB*TNN)          // 12288 token rows
#define TMLP 4096
#define TZ   256
#define TNA  32
#define TNG  24
#define TNP  20

// d_out flat layout (floats)
#define XR_OFF     0                       // 2048*144
#define LAT_OFF    (TB*144)                // 2048*256
#define LOGVAR_OFF (LAT_OFF + TB*TZ)       // 1
#define LOSS_OFF   (LOGVAR_OFF + 1)        // 1
#define PRED_OFF   (LOSS_OFF + 1)          // 2048
#define SP_OFF     (PRED_OFF + TB)         // 1

static inline int cdiv(int a, int b) { return (a + b - 1) / b; }

// ---------------------------------------------------------------------------
// WMMA GEMM: C[M,N] = epilogue( A[M,K] (f32) @ B[K,N] (f32) )
// 128x128x32 tiles, 8 wave32 per block, f16 WMMA / f32 accumulate.
// FAST=true  : M%128==0, N%128==0, K%32==0 -> no guards, double-buffered LDS
//              with register prefetch of the next K-tile.
// FAST=false : fully guarded single-buffer path for edge shapes.
// act: 0 = none, 1 = exact GELU (erf), 2 = SiLU.  Optional bias + residual.
// ---------------------------------------------------------------------------
template <bool FAST>
__global__ __launch_bounds__(256) void gemm_wmma_kernel(
    const float* __restrict__ A, const float* __restrict__ Bw,
    float* __restrict__ C, const float* __restrict__ bias,
    const float* __restrict__ Rres,
    int M, int N, int K, int ldC, int ldR, int act)
{
  __shared__ __align__(16) _Float16 As[2][128 * 40];  // M-major, 32 K + 8 pad
  __shared__ __align__(16) _Float16 Bs[2][128 * 40];  // N-major (B transposed)

  const int tid  = threadIdx.x;
  const int m0   = blockIdx.y * 128;
  const int n0   = blockIdx.x * 128;
  const int w    = tid >> 5;
  const int lane = tid & 31;
  const int ll   = lane & 15;
  const int hi   = lane >> 4;
  const int wm   = (w & 3) * 32;   // 4 waves along M
  const int wn   = (w >> 2) * 64;  // 2 waves along N

  v8f acc[2][4];
  const v8f vz = {0.f, 0.f, 0.f, 0.f, 0.f, 0.f, 0.f, 0.f};
  for (int i = 0; i < 2; i++)
    for (int j = 0; j < 4; j++) acc[i][j] = vz;

  if (FAST) {
    float4 ra[4], rb[4];
    // ---- prologue: global loads for tile 0 ----
#pragma unroll
    for (int i = 0; i < 4; i++) {
      int idx = tid + i * 256;
      ra[i] = *(const float4*)(A  + (size_t)(m0 + (idx >> 3)) * K + ((idx & 7) << 2));
      rb[i] = *(const float4*)(Bw + (size_t)(idx >> 5) * N + n0 + ((idx & 31) << 2));
    }
#pragma unroll
    for (int i = 0; i < 4; i++) {
      int idx = tid + i * 256;
      int r = idx >> 3, c4 = (idx & 7) << 2;
      v4h hv = {(_Float16)ra[i].x, (_Float16)ra[i].y, (_Float16)ra[i].z, (_Float16)ra[i].w};
      *(v4h*)&As[0][r * 40 + c4] = hv;
      int k = idx >> 5, cb = (idx & 31) << 2;
      Bs[0][(cb + 0) * 40 + k] = (_Float16)rb[i].x;
      Bs[0][(cb + 1) * 40 + k] = (_Float16)rb[i].y;
      Bs[0][(cb + 2) * 40 + k] = (_Float16)rb[i].z;
      Bs[0][(cb + 3) * 40 + k] = (_Float16)rb[i].w;
    }
    __syncthreads();

    const int nt = K >> 5;
    for (int t = 0; t < nt; t++) {
      const int cur = t & 1, nxt = cur ^ 1;
      // ---- prefetch next K-tile to registers (latency hidden by WMMAs) ----
      if (t + 1 < nt) {
        const int kk = (t + 1) << 5;
#pragma unroll
        for (int i = 0; i < 4; i++) {
          int idx = tid + i * 256;
          ra[i] = *(const float4*)(A  + (size_t)(m0 + (idx >> 3)) * K + kk + ((idx & 7) << 2));
          rb[i] = *(const float4*)(Bw + (size_t)(kk + (idx >> 5)) * N + n0 + ((idx & 31) << 2));
        }
      }
      // ---- fragments (16-bit 16x32 A layout) + 8 WMMAs ----
      v16h af[2], bf[4];
#pragma unroll
      for (int fi = 0; fi < 2; fi++) {
        const _Float16* p = &As[cur][(wm + fi * 16 + ll) * 40 + hi * 8];
        v8h lo  = *(const v8h*)p;
        v8h hi8 = *(const v8h*)(p + 16);
        af[fi] = __builtin_shufflevector(lo, hi8, 0, 1, 2, 3, 4, 5, 6, 7,
                                         8, 9, 10, 11, 12, 13, 14, 15);
      }
#pragma unroll
      for (int fj = 0; fj < 4; fj++) {
        const _Float16* p = &Bs[cur][(wn + fj * 16 + ll) * 40 + hi * 8];
        v8h lo  = *(const v8h*)p;
        v8h hi8 = *(const v8h*)(p + 16);
        bf[fj] = __builtin_shufflevector(lo, hi8, 0, 1, 2, 3, 4, 5, 6, 7,
                                         8, 9, 10, 11, 12, 13, 14, 15);
      }
#pragma unroll
      for (int fi = 0; fi < 2; fi++)
#pragma unroll
        for (int fj = 0; fj < 4; fj++)
          acc[fi][fj] = __builtin_amdgcn_wmma_f32_16x16x32_f16(
              false, af[fi], false, bf[fj], (short)0, acc[fi][fj], false, false);
      // ---- store prefetched tile into the other buffer ----
      if (t + 1 < nt) {
#pragma unroll
        for (int i = 0; i < 4; i++) {
          int idx = tid + i * 256;
          int r = idx >> 3, c4 = (idx & 7) << 2;
          v4h hv = {(_Float16)ra[i].x, (_Float16)ra[i].y, (_Float16)ra[i].z, (_Float16)ra[i].w};
          *(v4h*)&As[nxt][r * 40 + c4] = hv;
          int k = idx >> 5, cb = (idx & 31) << 2;
          Bs[nxt][(cb + 0) * 40 + k] = (_Float16)rb[i].x;
          Bs[nxt][(cb + 1) * 40 + k] = (_Float16)rb[i].y;
          Bs[nxt][(cb + 2) * 40 + k] = (_Float16)rb[i].z;
          Bs[nxt][(cb + 3) * 40 + k] = (_Float16)rb[i].w;
        }
      }
      __syncthreads();
    }
  } else {
    // ------------------ guarded generic path (edge shapes) ------------------
    for (int kk = 0; kk < K; kk += 32) {
      const bool fullK = (kk + 32) <= K;
#pragma unroll
      for (int i = 0; i < 4; i++) {
        int idx = tid + i * 256;
        int r   = idx >> 3;
        int c4  = (idx & 7) << 2;
        int gr  = m0 + r;
        float x0 = 0.f, x1 = 0.f, x2 = 0.f, x3 = 0.f;
        if (gr < M) {
          if (fullK) {
            const float4 f = *(const float4*)(A + (size_t)gr * K + kk + c4);
            x0 = f.x; x1 = f.y; x2 = f.z; x3 = f.w;
          } else {
            int kb = kk + c4;
            if (kb + 0 < K) x0 = A[(size_t)gr * K + kb + 0];
            if (kb + 1 < K) x1 = A[(size_t)gr * K + kb + 1];
            if (kb + 2 < K) x2 = A[(size_t)gr * K + kb + 2];
            if (kb + 3 < K) x3 = A[(size_t)gr * K + kb + 3];
          }
        }
        v4h hv = {(_Float16)x0, (_Float16)x1, (_Float16)x2, (_Float16)x3};
        *(v4h*)&As[0][r * 40 + c4] = hv;
      }
#pragma unroll
      for (int i = 0; i < 4; i++) {
        int idx = tid + i * 256;
        int k   = idx >> 5;
        int c4  = (idx & 31) << 2;
        int gk  = kk + k;
        float x0 = 0.f, x1 = 0.f, x2 = 0.f, x3 = 0.f;
        if (gk < K) {
          int gn = n0 + c4;
          if (gn + 3 < N) {
            const float4 f = *(const float4*)(Bw + (size_t)gk * N + gn);
            x0 = f.x; x1 = f.y; x2 = f.z; x3 = f.w;
          } else {
            if (gn + 0 < N) x0 = Bw[(size_t)gk * N + gn + 0];
            if (gn + 1 < N) x1 = Bw[(size_t)gk * N + gn + 1];
            if (gn + 2 < N) x2 = Bw[(size_t)gk * N + gn + 2];
            if (gn + 3 < N) x3 = Bw[(size_t)gk * N + gn + 3];
          }
        }
        Bs[0][(c4 + 0) * 40 + k] = (_Float16)x0;
        Bs[0][(c4 + 1) * 40 + k] = (_Float16)x1;
        Bs[0][(c4 + 2) * 40 + k] = (_Float16)x2;
        Bs[0][(c4 + 3) * 40 + k] = (_Float16)x3;
      }
      __syncthreads();

      v16h af[2], bf[4];
#pragma unroll
      for (int fi = 0; fi < 2; fi++) {
        const _Float16* p = &As[0][(wm + fi * 16 + ll) * 40 + hi * 8];
        v8h lo  = *(const v8h*)p;
        v8h hi8 = *(const v8h*)(p + 16);
        af[fi] = __builtin_shufflevector(lo, hi8, 0, 1, 2, 3, 4, 5, 6, 7,
                                         8, 9, 10, 11, 12, 13, 14, 15);
      }
#pragma unroll
      for (int fj = 0; fj < 4; fj++) {
        const _Float16* p = &Bs[0][(wn + fj * 16 + ll) * 40 + hi * 8];
        v8h lo  = *(const v8h*)p;
        v8h hi8 = *(const v8h*)(p + 16);
        bf[fj] = __builtin_shufflevector(lo, hi8, 0, 1, 2, 3, 4, 5, 6, 7,
                                         8, 9, 10, 11, 12, 13, 14, 15);
      }
#pragma unroll
      for (int fi = 0; fi < 2; fi++)
#pragma unroll
        for (int fj = 0; fj < 4; fj++)
          acc[fi][fj] = __builtin_amdgcn_wmma_f32_16x16x32_f16(
              false, af[fi], false, bf[fj], (short)0, acc[fi][fj], false, false);
      __syncthreads();
    }
  }

  // ---- epilogue: bias + activation + residual, f32 store ----
#pragma unroll
  for (int fi = 0; fi < 2; fi++) {
#pragma unroll
    for (int fj = 0; fj < 4; fj++) {
      int gcol = n0 + wn + fj * 16 + ll;
      if (!FAST && gcol >= N) continue;
      float bv = bias ? bias[gcol] : 0.f;
#pragma unroll
      for (int r = 0; r < 8; r++) {
        int grow = m0 + wm + fi * 16 + hi * 8 + r;
        if (!FAST && grow >= M) continue;
        float v = acc[fi][fj][r] + bv;
        if (act == 1)       v = 0.5f * v * (1.f + erff(v * 0.70710678118f));
        else if (act == 2)  v = v * (1.f / (1.f + __expf(-v)));
        if (Rres) v += Rres[(size_t)grow * ldR + gcol];
        C[(size_t)grow * ldC + gcol] = v;
      }
    }
  }
}

// ---------------------------------------------------------------------------
// LayerNorm over D=1024, one block (256 thr) per row.
// ---------------------------------------------------------------------------
__global__ __launch_bounds__(256) void ln_kernel(
    const float* __restrict__ X, float* __restrict__ Y,
    const float* __restrict__ g, const float* __restrict__ b)
{
  __shared__ float sh[256], sh2[256];
  const int row = blockIdx.x, tid = threadIdx.x;
  const float* xr = X + (size_t)row * TD;
  float s = 0.f, s2 = 0.f;
  for (int i = tid; i < TD; i += 256) { float v = xr[i]; s += v; s2 += v * v; }
  sh[tid] = s; sh2[tid] = s2; __syncthreads();
  for (int off = 128; off > 0; off >>= 1) {
    if (tid < off) { sh[tid] += sh[tid + off]; sh2[tid] += sh2[tid + off]; }
    __syncthreads();
  }
  const float mu  = sh[0] * (1.f / TD);
  const float var = sh2[0] * (1.f / TD) - mu * mu;
  const float rs  = rsqrtf(var + 1e-5f);
  float* yr = Y + (size_t)row * TD;
  for (int i = tid; i < TD; i += 256)
    yr[i] = (xr[i] - mu) * rs * g[i] + b[i];
}

// ---------------------------------------------------------------------------
// Attention (N=6, head dim 256) + alibi + entropy accumulation.
// One thread per (batch, head, query-row); reads qkv in place.
// ---------------------------------------------------------------------------
__global__ __launch_bounds__(256) void attn_kernel(
    const float* __restrict__ qkv, const float* __restrict__ alibi,
    float* __restrict__ out, float* __restrict__ ent_acc)
{
  int t = blockIdx.x * 256 + threadIdx.x;
  if (t >= TB * THH * TNN) return;
  const int i  = t % TNN;
  const int hh = (t / TNN) % THH;
  const int b  = t / (TNN * THH);

  const float* q = qkv + ((size_t)(b * TNN + i)) * (3 * TD) + hh * TDH;
  float dots[TNN];
  for (int j = 0; j < TNN; j++) {
    const float* k = qkv + ((size_t)(b * TNN + j)) * (3 * TD) + TD + hh * TDH;
    float d = 0.f;
    for (int dd = 0; dd < TDH; dd++) d += q[dd] * k[dd];
    dots[j] = d * 0.0625f + alibi[b * 36 + i * TNN + j];
  }
  float mx = dots[0];
  for (int j = 1; j < TNN; j++) mx = fmaxf(mx, dots[j]);
  float se = 0.f;
  for (int j = 0; j < TNN; j++) { dots[j] = __expf(dots[j] - mx); se += dots[j]; }
  const float inv = 1.f / se;
  float ent = 0.f;
  for (int j = 0; j < TNN; j++) {
    float p = dots[j] * inv; dots[j] = p;
    ent -= p * __logf(p + 1e-8f);
  }
  atomicAdd(ent_acc, ent);

  float* o = out + ((size_t)(b * TNN + i)) * TD + hh * TDH;
  for (int dd = 0; dd < TDH; dd++) {
    float s = 0.f;
    for (int j = 0; j < TNN; j++)
      s += dots[j] * qkv[((size_t)(b * TNN + j)) * (3 * TD) + 2 * TD + hh * TDH + dd];
    o[dd] = s;
  }
}

// ---------------------------------------------------------------------------
// Elementwise helpers
// ---------------------------------------------------------------------------
__global__ void film_kernel(float* __restrict__ h, const float* __restrict__ fg,
                            const float* __restrict__ fb)
{
  size_t i = (size_t)blockIdx.x * 256 + threadIdx.x;
  if (i >= (size_t)TM * TD) return;
  int d = i & (TD - 1);
  int b = (int)((i >> 10) / TNN);
  h[i] = fg[(size_t)b * TD + d] * h[i] + fb[(size_t)b * TD + d];
}

__global__ void add_emb_kernel(float* __restrict__ h, const float* __restrict__ emb)
{
  size_t i = (size_t)blockIdx.x * 256 + threadIdx.x;
  if (i >= (size_t)TM * TD) return;
  int d = i & (TD - 1);
  int n = (int)((i >> 10) % TNN);
  h[i] += emb[n * TD + d];
}

__global__ void bcast_dec_kernel(const float* __restrict__ dl,
                                 const float* __restrict__ emb,
                                 float* __restrict__ h)
{
  size_t i = (size_t)blockIdx.x * 256 + threadIdx.x;
  if (i >= (size_t)TM * TD) return;
  int d  = i & (TD - 1);
  int bn = (int)(i >> 10);
  int b = bn / TNN, n = bn % TNN;
  h[i] = dl[(size_t)b * TD + d] + emb[n * TD + d];
}

__global__ void meanpool_kernel(const float* __restrict__ src, float* __restrict__ dst)
{
  size_t i = (size_t)blockIdx.x * 256 + threadIdx.x;
  if (i >= (size_t)TB * TD) return;
  int d = i & (TD - 1);
  int b = (int)(i >> 10);
  float s = 0.f;
  for (int n = 0; n < TNN; n++) s += src[((size_t)b * TNN + n) * TD + d];
  dst[i] = s * (1.f / TNN);
}

__global__ void prep_kernel(const float* __restrict__ x, const float* __restrict__ adm,
                            float* __restrict__ node, float* __restrict__ gp,
                            float* __restrict__ alibi, int* __restrict__ algid)
{
  int b = blockIdx.x * 256 + threadIdx.x;
  if (b >= TB) return;
  const float* xr = x + (size_t)b * 145;
  int alg = (int)xr[134];
  if (alg < 0) alg = 0;
  if (alg > TNA - 1) alg = TNA - 1;
  algid[b] = alg;
  for (int j = 0; j < TNN * TNP; j++) node[(size_t)b * (TNN * TNP) + j] = xr[j];
  for (int j = 0; j < 14; j++) gp[b * TNG + j] = xr[120 + j];
  for (int j = 0; j < 10; j++) gp[b * TNG + 14 + j] = xr[135 + j];
  for (int e = 0; e < 36; e++) alibi[b * 36 + e] = adm[alg * 36 + e];
}

__global__ void alg_kernel(const float* __restrict__ probs, const int* __restrict__ algid,
                           float* __restrict__ loss_acc, float* __restrict__ pred)
{
  int b = blockIdx.x * 256 + threadIdx.x;
  if (b >= TB) return;
  const float* p = probs + (size_t)b * TNA;
  float mx = p[0]; int am = 0;
  for (int j = 1; j < TNA; j++)
    if (p[j] > mx) { mx = p[j]; am = j; }
  float se = 0.f;
  for (int j = 0; j < TNA; j++) se += __expf(p[j] - mx);
  float lse = mx + __logf(se);
  atomicAdd(loss_acc, lse - p[algid[b]]);
  pred[b] = (float)am;
}

__global__ void copy_ghat_kernel(const float* __restrict__ ghat, float* __restrict__ out)
{
  int i = blockIdx.x * 256 + threadIdx.x;
  if (i >= TB * TNG) return;
  int b = i / TNG, j = i % TNG;
  out[(size_t)b * 144 + 120 + j] = ghat[i];
}

__global__ void copy_recon_kernel(const float* __restrict__ rec, float* __restrict__ out)
{
  int i = blockIdx.x * 256 + threadIdx.x;
  if (i >= TM * TNP) return;
  int row = i / TNP, c = i % TNP;
  int b = row / TNN, n = row % TNN;
  out[(size_t)b * 144 + n * TNP + c] = rec[i];
}

__global__ void init_sc_kernel(float* sc)
{
  if (threadIdx.x < 8) sc[threadIdx.x] = 0.f;
}

__global__ void finalize_kernel(const float* __restrict__ sc, float* __restrict__ out)
{
  out[LOGVAR_OFF] = 1.0f;
  out[LOSS_OFF]   = sc[1] * (1.f / TB);
  out[SP_OFF]     = sc[0] * 0.01f / (float)(TB * THH * TNN);
}

// ---------------------------------------------------------------------------
// Host-side orchestration
// ---------------------------------------------------------------------------
static void gemm(hipStream_t st, const float* A, const float* B, float* C,
                 const float* bias, const float* R,
                 int M, int N, int K, int ldC, int ldR, int act)
{
  dim3 g(cdiv(N, 128), cdiv(M, 128));
  const bool fast = (M % 128 == 0) && (N % 128 == 0) && (K % 32 == 0);
  if (fast)
    gemm_wmma_kernel<true><<<g, 256, 0, st>>>(A, B, C, bias, R, M, N, K, ldC, ldR, act);
  else
    gemm_wmma_kernel<false><<<g, 256, 0, st>>>(A, B, C, bias, R, M, N, K, ldC, ldR, act);
}

struct TParams {
  const float *ln1g, *ln1b, *wqkv, *wout, *ln2g, *ln2b, *w1, *b1, *w2, *b2, *ng, *nb;
};

static void run_transformer(hipStream_t st, const TParams& P,
                            float* H, float* T1, float* T2,
                            const float* FG, const float* FB,
                            const float* ALIBI, float* ent_acc)
{
  const int ew_grid = cdiv(TM * TD, 256);
  for (int i = 0; i < 6; i++) {
    // attention block
    ln_kernel<<<TM, 256, 0, st>>>(H, T2, P.ln1g + (size_t)i * TD, P.ln1b + (size_t)i * TD);
    gemm(st, T2, P.wqkv + (size_t)i * TD * 3 * TD, T1, nullptr, nullptr,
         TM, 3 * TD, TD, 3 * TD, 0, 0);
    attn_kernel<<<cdiv(TB * THH * TNN, 256), 256, 0, st>>>(T1, ALIBI, T2, ent_acc);
    gemm(st, T2, P.wout + (size_t)i * TD * TD, H, nullptr, H,
         TM, TD, TD, TD, TD, 0);                              // h = attn_proj + h
    film_kernel<<<ew_grid, 256, 0, st>>>(H, FG, FB);
    // FF block
    ln_kernel<<<TM, 256, 0, st>>>(H, T2, P.ln2g + (size_t)i * TD, P.ln2b + (size_t)i * TD);
    gemm(st, T2, P.w1 + (size_t)i * TD * TMLP, T1, P.b1 + (size_t)i * TMLP, nullptr,
         TM, TMLP, TD, TMLP, 0, 1);                           // exact GELU
    gemm(st, T1, P.w2 + (size_t)i * TMLP * TD, H, P.b2 + (size_t)i * TD, H,
         TM, TD, TMLP, TD, TD, 0);                            // h = ff + h
  }
  ln_kernel<<<TM, 256, 0, st>>>(H, T2, P.ng, P.nb);           // final LN -> T2
}

extern "C" void kernel_launch(void* const* d_in, const int* in_sizes, int n_in,
                              void* d_out, int out_size, void* d_ws, size_t ws_size,
                              hipStream_t stream)
{
  (void)in_sizes; (void)n_in; (void)out_size; (void)ws_size;

  // ---- inputs (setup_inputs() dict order, depth-first) ----
  const float* x = (const float*)d_in[0];
  TParams enc = {
    (const float*)d_in[1],  (const float*)d_in[2],  (const float*)d_in[3],
    (const float*)d_in[4],  (const float*)d_in[5],  (const float*)d_in[6],
    (const float*)d_in[7],  (const float*)d_in[8],  (const float*)d_in[9],
    (const float*)d_in[10], (const float*)d_in[11], (const float*)d_in[12]};
  TParams dec = {
    (const float*)d_in[13], (const float*)d_in[14], (const float*)d_in[15],
    (const float*)d_in[16], (const float*)d_in[17], (const float*)d_in[18],
    (const float*)d_in[19], (const float*)d_in[20], (const float*)d_in[21],
    (const float*)d_in[22], (const float*)d_in[23], (const float*)d_in[24]};
  const float* W_in   = (const float*)d_in[25];
  const float* b_in   = (const float*)d_in[26];
  const float* emb    = (const float*)d_in[27];
  const float* adm    = (const float*)d_in[28];
  const float* ge_W1  = (const float*)d_in[29];
  const float* ge_b1  = (const float*)d_in[30];
  const float* ge_W2  = (const float*)d_in[31];
  const float* ge_b2  = (const float*)d_in[32];
  const float* encg_W = (const float*)d_in[33];
  const float* encg_b = (const float*)d_in[34];
  const float* encb_W = (const float*)d_in[35];
  const float* encb_b = (const float*)d_in[36];
  const float* decg_W = (const float*)d_in[37];
  const float* decg_b = (const float*)d_in[38];
  const float* decb_W = (const float*)d_in[39];
  const float* decb_b = (const float*)d_in[40];
  const float* lat_W  = (const float*)d_in[41];
  const float* lat_b  = (const float*)d_in[42];
  const float* flat_W = (const float*)d_in[43];
  const float* flat_b = (const float*)d_in[44];
  const float* alg_W  = (const float*)d_in[45];
  const float* alg_b  = (const float*)d_in[46];
  const float* gp_W1  = (const float*)d_in[47];
  const float* gp_b1  = (const float*)d_in[48];
  const float* gp_W2  = (const float*)d_in[49];
  const float* gp_b2  = (const float*)d_in[50];
  const float* out_W  = (const float*)d_in[51];
  const float* out_b  = (const float*)d_in[52];

  // ---- workspace layout (floats) ----
  float* ws = (float*)d_ws;
  size_t o = 0;
  float* H     = ws + o; o += (size_t)TM * TD;     // persistent hidden state
  float* T1    = ws + o; o += (size_t)TM * TMLP;   // qkv / FF hidden scratch
  float* T2    = ws + o; o += (size_t)TM * TD;     // LN / attn-out scratch
  float* GE    = ws + o; o += (size_t)TB * TD;     // graph embedding
  float* GEB   = ws + o; o += (size_t)TB * TD;     // gate intermediate
  float* FG    = ws + o; o += (size_t)TB * TD;     // film gamma
  float* FB    = ws + o; o += (size_t)TB * TD;     // film beta
  float* POOL  = ws + o; o += (size_t)TB * TD;     // pooled / latent-broadcast src
  float* GHAT  = ws + o; o += (size_t)TB * TNG;
  float* GP    = ws + o; o += (size_t)TB * TNG;
  float* ALIBI = ws + o; o += (size_t)TB * 36;
  float* PROBS = ws + o; o += (size_t)TB * TNA;
  float* RECON = ws + o; o += (size_t)TM * TNP;
  float* NODE  = ws + o; o += (size_t)TM * TNP;
  float* SC    = ws + o; o += 8;                   // SC[0]=entropy sum, SC[1]=loss sum
  int*   ALGID = (int*)(ws + o); o += TB;

  float* outf = (float*)d_out;
  float* LAT  = outf + LAT_OFF;                    // latent lives directly in d_out
  const int ew_grid = cdiv(TM * TD, 256);

  // ---- front end ----
  init_sc_kernel<<<1, 32, 0, stream>>>(SC);
  prep_kernel<<<cdiv(TB, 256), 256, 0, stream>>>(x, adm, NODE, GP, ALIBI, ALGID);

  gemm(stream, NODE, W_in, H, b_in, nullptr, TM, TD, TNP, TD, 0, 0);
  add_emb_kernel<<<ew_grid, 256, 0, stream>>>(H, emb);

  gemm(stream, GP,  ge_W1, GEB, ge_b1, nullptr, TB, TD, TNG, TD, 0, 2);  // silu
  gemm(stream, GEB, ge_W2, GE,  ge_b2, nullptr, TB, TD, TD,  TD, 0, 0);
  gemm(stream, GE, encg_W, FG, encg_b, nullptr, TB, TD, TD, TD, 0, 2);
  gemm(stream, GE, encb_W, FB, encb_b, nullptr, TB, TD, TD, TD, 0, 2);

  // ---- encoder ----
  run_transformer(stream, enc, H, T1, T2, FG, FB, ALIBI, SC);

  // ---- latent heads ----
  meanpool_kernel<<<cdiv(TB * TD, 256), 256, 0, stream>>>(T2, POOL);
  gemm(stream, POOL, lat_W, LAT, lat_b, nullptr, TB, TZ, TD, TZ, 0, 0);
  gemm(stream, LAT, alg_W, PROBS, alg_b, nullptr, TB, TNA, TZ, TNA, 0, 0);
  alg_kernel<<<cdiv(TB, 256), 256, 0, stream>>>(PROBS, ALGID, SC + 1, outf + PRED_OFF);

  gemm(stream, LAT, gp_W1, GEB, gp_b1, nullptr, TB, TD, TZ, TD, 0, 2);
  gemm(stream, GEB, gp_W2, GHAT, gp_b2, nullptr, TB, TNG, TD, TNG, 0, 0);
  copy_ghat_kernel<<<cdiv(TB * TNG, 256), 256, 0, stream>>>(GHAT, outf);

  // ---- decoder conditioning ----
  gemm(stream, GHAT, ge_W1, GEB, ge_b1, nullptr, TB, TD, TNG, TD, 0, 2);
  gemm(stream, GEB,  ge_W2, GE,  ge_b2, nullptr, TB, TD, TD,  TD, 0, 0);
  gemm(stream, GE, decg_W, FG, decg_b, nullptr, TB, TD, TD, TD, 0, 2);
  gemm(stream, GE, decb_W, FB, decb_b, nullptr, TB, TD, TD, TD, 0, 2);

  gemm(stream, LAT, flat_W, POOL, flat_b, nullptr, TB, TD, TZ, TD, 0, 0);
  bcast_dec_kernel<<<ew_grid, 256, 0, stream>>>(POOL, emb, H);

  // ---- decoder ----
  run_transformer(stream, dec, H, T1, T2, FG, FB, ALIBI, SC);

  // ---- outputs ----
  gemm(stream, T2, out_W, RECON, out_b, nullptr, TM, TNP, TD, TNP, 0, 0);
  copy_recon_kernel<<<cdiv(TM * TNP, 256), 256, 0, stream>>>(RECON, outf);
  finalize_kernel<<<1, 1, 0, stream>>>(SC, outf);
}
